// PointHead_59021440582019
// MI455X (gfx1250) — matrix operand order repaired and checked by reading, not compiled, not run on hardware
//
#include <hip/hip_runtime.h>
#include <hip/hip_bf16.h>
#include <stdint.h>

typedef __attribute__((ext_vector_type(16))) __bf16 v16bf;
typedef __attribute__((ext_vector_type(8)))  __bf16 v8bf;
typedef __attribute__((ext_vector_type(8)))  float  v8f;

#define BATCH 16
#define HC 60
#define WC 80
#define HF 480
#define WF 640
#define NPIX (BATCH*HC*WC)      /* 76800 */
#define NFULL (BATCH*HF*WF)     /* 4915200 */

__device__ inline v8f wmma_bf16(v16bf a, v16bf b, v8f c) {
  return __builtin_amdgcn_wmma_f32_16x16x32_bf16(false, a, false, b, (short)0, c,
                                                 false, false);
}

// ---------------- layout transform: NCHW f32 -> NHWC bf16 ----------------
__global__ void nchw_to_nhwc_bf16_kernel(const float* __restrict__ in,
                                         __bf16* __restrict__ out,
                                         int C, int H, int W, int total) {
  int idx = blockIdx.x * blockDim.x + threadIdx.x;
  if (idx >= total) return;
  int c = idx % C;
  int t = idx / C;
  int w = t % W; t /= W;
  int h = t % H;
  int b = t / H;
  out[idx] = (__bf16)in[(((size_t)b * C + c) * H + h) * W + w];
}

// ---------------- weight repack into WMMA A-fragment order ----------------
// packed element: tile = ((tap*nCb + cb)*nOt + ot), within-tile = (m*2+hl)*16 + e
// value = w[o = ot*16+m, c = cb*32 + K(e,hl), tap]   (zero-padded)
__global__ void pack_weights_kernel(const float* __restrict__ w,
                                    __bf16* __restrict__ wp,
                                    int O, int Cin, int taps, int nCb, int nOt,
                                    int total) {
  int idx = blockIdx.x * blockDim.x + threadIdx.x;
  if (idx >= total) return;
  int e  = idx & 15;
  int mh = (idx >> 4) & 31;
  int m  = mh >> 1;
  int hl = mh & 1;
  int tile = idx >> 9;
  int ot = tile % nOt;
  int r  = tile / nOt;
  int cb = r % nCb;
  int tap = r / nCb;
  int o = ot * 16 + m;
  int k = e + 16 * hl;
  int c = cb * 32 + k;
  float val = 0.f;
  if (o < O && c < Cin) val = w[((size_t)o * Cin + c) * taps + tap];
  wp[idx] = (__bf16)val;
}

// ---------------- async staging: one K-block of input rows into LDS ----------
// Writes 32 channels x 3 rows x pixels w=-1..80 (in-bounds entries only; OOB
// halo entries remain at the zeros written once at kernel start).
template <int CIN>
__device__ inline void stage_async(const __bf16* __restrict__ in,
                                   __bf16 (*buf)[88][32],
                                   int b, int h, int cb, int tid) {
  for (int i = tid; i < 3 * 82 * 4; i += 256) {
    int chunk = i & 3;
    int pix = i >> 2;
    int r  = pix / 82;
    int px = pix % 82;
    int hh = h + r - 1;
    int ww = px - 1;
    if (hh >= 0 && hh < HC && ww >= 0 && ww < WC) {
      unsigned lds = (unsigned)(uintptr_t)(&buf[r][px][chunk * 8]);
      const __bf16* g = in + (((size_t)(b * HC + hh) * WC + ww) * CIN +
                              cb * 32 + chunk * 8);
      // ASYNCcnt-tracked global->LDS DMA (no VGPR data round-trip)
      asm volatile("global_load_async_to_lds_b128 %0, %1, off"
                   :: "v"(lds), "v"((unsigned long long)(uintptr_t)g)
                   : "memory");
    }
  }
}

__device__ inline void wait_async_all() {
  asm volatile("s_wait_asynccnt 0x0" ::: "memory");
}

// ---------------- 3x3 conv, implicit GEMM with bf16 WMMA ----------------
// workgroup: 8 waves, computes one (b,h) output row: COUT channels x 80 pixels
// Double-buffered LDS staging (async DMA) overlaps K-block cb+1 with the
// 9-tap WMMA sweep over cb; A fragments are register-double-buffered across
// taps so weight-load latency is hidden behind a full tap of WMMAs.
template <int CIN, int COUT, int MT>
__global__ void __launch_bounds__(256)
conv3x3_wmma(const __bf16* __restrict__ in,   // NHWC bf16 [B,HC,WC,CIN]
             const __bf16* __restrict__ wp,   // packed weights
             const float* __restrict__ bias,
             __bf16* __restrict__ out) {      // NHWC bf16 [B,HC,WC,COUT]
  constexpr int nCb = CIN / 32;
  constexpr int nOt = COUT / 16;
  const int h = blockIdx.x;
  const int b = blockIdx.y;
  const int tid  = threadIdx.x;
  const int wave = tid >> 5;
  const int lane = tid & 31;
  const int hl = lane >> 4;
  const int nn = lane & 15;

  __shared__ __attribute__((aligned(32))) __bf16 sIn[2][3][88][32];

  v8f acc[MT][5];
#pragma unroll
  for (int mt = 0; mt < MT; ++mt)
#pragma unroll
    for (int nt = 0; nt < 5; ++nt)
#pragma unroll
      for (int k = 0; k < 8; ++k) acc[mt][nt][k] = 0.f;

  // zero both buffers once: halo entries are never touched by async loads
  {
    uint4 z = {0u, 0u, 0u, 0u};
    uint4* p = (uint4*)(&sIn[0][0][0][0]);
    for (int i = tid; i < 2 * 3 * 88 * 32 * 2 / 16; i += 256) p[i] = z;
  }
  __syncthreads();

  stage_async<CIN>(in, sIn[0], b, h, 0, tid);
  wait_async_all();
  __syncthreads();

  auto load_afrag = [&](int tap, int cb, v16bf* dst) {
#pragma unroll
    for (int mt = 0; mt < MT; ++mt) {
      int ot = wave * MT + mt;
      size_t base = ((size_t)((tap * nCb + cb) * nOt + ot)) * 512 +
                    (size_t)nn * 32 + (size_t)hl * 16;
      dst[mt] = *(const v16bf*)(wp + base);
    }
  };

  int buf = 0;
  for (int cb = 0; cb < nCb; ++cb) {
    // kick off DMA for the next K-block while computing this one
    if (cb + 1 < nCb) stage_async<CIN>(in, sIn[buf ^ 1], b, h, cb + 1, tid);

    v16bf aA[MT], aB[MT];
    load_afrag(0, cb, aA);

#pragma unroll
    for (int tap = 0; tap < 9; ++tap) {
      const int ky = tap / 3, kx = tap % 3;
      v16bf* curA = (tap & 1) ? aB : aA;
      v16bf* nxtA = (tap & 1) ? aA : aB;
      // prefetch next tap's weights before this tap's WMMAs
      if (tap + 1 < 9) load_afrag(tap + 1, cb, nxtA);
#pragma unroll
      for (int nt = 0; nt < 5; ++nt) {
        int px = nt * 16 + nn + kx;  // w+kx-1 shifted by +1 pad
        v16bf bfrag = *(const v16bf*)(&sIn[buf][ky][px][16 * hl]);
#pragma unroll
        for (int mt = 0; mt < MT; ++mt)
          acc[mt][nt] = wmma_bf16(curA[mt], bfrag, acc[mt][nt]);
      }
    }

    wait_async_all();   // next buffer's DMA complete (per-wave ASYNCcnt)
    __syncthreads();    // all waves done reading old buffer / next one ready
    buf ^= 1;
  }

  // epilogue: bias + relu + cvt bf16, NHWC store (8 consecutive channels/lane)
#pragma unroll
  for (int mt = 0; mt < MT; ++mt) {
    int ot = wave * MT + mt;
    int o0 = ot * 16 + hl * 8;
    float bb[8];
#pragma unroll
    for (int v = 0; v < 8; ++v) bb[v] = bias[o0 + v];
#pragma unroll
    for (int nt = 0; nt < 5; ++nt) {
      int w = nt * 16 + nn;
      union { v8bf v; uint4 u; } pk;
#pragma unroll
      for (int v = 0; v < 8; ++v) {
        float r = acc[mt][nt][v] + bb[v];
        r = fmaxf(r, 0.f);
        pk.v[v] = (__bf16)r;
      }
      *(uint4*)(out + (((size_t)(b * HC + h) * WC + w) * COUT + o0)) = pk.u;
    }
  }
}

// ---------------- 1x1 conv (65 -> padded 80 outs), K=128, WMMA ----------------
// 5 waves; wave = n-tile of 16 pixels; each wave does 5 M-tiles over K=128.
// A fragments register-double-buffered across the 4 K-blocks.
__global__ void __launch_bounds__(160)
conv1x1_wmma(const __bf16* __restrict__ in,  // NHWC bf16 [B,HC,WC,128]
             const __bf16* __restrict__ wp,  // packed [4 cb][5 ot][512]
             const float* __restrict__ bias, // 65 entries
             float* __restrict__ out) {      // NHWC f32 [B,HC,WC,80]
  const int h = blockIdx.x;
  const int b = blockIdx.y;
  const int tid  = threadIdx.x;
  const int wave = tid >> 5;
  const int lane = tid & 31;
  const int hl = lane >> 4;
  const int nn = lane & 15;
  const int w = wave * 16 + nn;

  v8f acc[5];
#pragma unroll
  for (int mt = 0; mt < 5; ++mt)
#pragma unroll
    for (int k = 0; k < 8; ++k) acc[mt][k] = 0.f;

  // hoist all B fragments: 4x contiguous 32B loads clause together
  const __bf16* pin = in + ((size_t)(b * HC + h) * WC + w) * 128;
  v16bf bfr[4];
#pragma unroll
  for (int kb = 0; kb < 4; ++kb)
    bfr[kb] = *(const v16bf*)(pin + kb * 32 + 16 * hl);

  auto load_a = [&](int kb, v16bf* dst) {
#pragma unroll
    for (int mt = 0; mt < 5; ++mt) {
      size_t base = ((size_t)(kb * 5 + mt)) * 512 + (size_t)nn * 32 + (size_t)hl * 16;
      dst[mt] = *(const v16bf*)(wp + base);
    }
  };

  v16bf aA[5], aB[5];
  load_a(0, aA);
#pragma unroll
  for (int kb = 0; kb < 4; ++kb) {
    v16bf* cur = (kb & 1) ? aB : aA;
    v16bf* nxt = (kb & 1) ? aA : aB;
    if (kb + 1 < 4) load_a(kb + 1, nxt);   // prefetch next K-block weights
#pragma unroll
    for (int mt = 0; mt < 5; ++mt)
      acc[mt] = wmma_bf16(cur[mt], bfr[kb], acc[mt]);
  }

  float* po = out + ((size_t)(b * HC + h) * WC + w) * 80;
#pragma unroll
  for (int mt = 0; mt < 5; ++mt) {
    int o0 = mt * 16 + hl * 8;
#pragma unroll
    for (int v = 0; v < 8; ++v) {
      int o = o0 + v;
      float bv = (o < 65) ? bias[o] : 0.f;
      po[o] = acc[mt][v] + bv;
    }
  }
}

// ---------------- softmax over 65 channels + pixel unshuffle ----------------
__global__ void softmax_unshuffle_kernel(const float* __restrict__ logits,
                                         float* __restrict__ S) {
  int idx = blockIdx.x * blockDim.x + threadIdx.x;
  if (idx >= NPIX) return;
  int w = idx % WC;
  int t = idx / WC;
  int h = t % HC;
  int b = t / HC;
  const float* p = logits + (size_t)idx * 80;
  float mx = p[0];
  for (int c = 1; c < 65; ++c) mx = fmaxf(mx, p[c]);
  float e[64];
  float sum = 0.f;
  for (int c = 0; c < 65; ++c) {
    float v = __expf(p[c] - mx);
    if (c < 64) e[c] = v;
    sum += v;
  }
  float inv = 1.f / sum;
  for (int c = 0; c < 64; ++c) {
    int yy = h * 8 + (c >> 3);
    int xx = w * 8 + (c & 7);
    S[((size_t)b * HF + yy) * WF + xx] = e[c] * inv;
  }
}

// ---------------- separable 9-tap max pools + NMS elementwise ----------------
__global__ void maxpool_h9_kernel(const float* __restrict__ in, float* __restrict__ out) {
  int idx = blockIdx.x * blockDim.x + threadIdx.x;
  if (idx >= NFULL) return;
  int x = idx % WF;
  int rowbase = idx - x;
  float m = in[idx];
  for (int d = -4; d <= 4; ++d) {
    int xx = x + d;
    if (xx >= 0 && xx < WF) m = fmaxf(m, in[rowbase + xx]);
  }
  out[idx] = m;
}

__global__ void maxpool_v9_kernel(const float* __restrict__ in, float* __restrict__ out) {
  int idx = blockIdx.x * blockDim.x + threadIdx.x;
  if (idx >= NFULL) return;
  int x = idx % WF;
  int t = idx / WF;
  int y = t % HF;
  int b = t / HF;
  const float* plane = in + (size_t)b * HF * WF;
  float m = in[idx];
  for (int d = -4; d <= 4; ++d) {
    int yy = y + d;
    if (yy >= 0 && yy < HF) m = fmaxf(m, plane[yy * WF + x]);
  }
  out[idx] = m;
}

__global__ void eq_mask_kernel(const float* __restrict__ S, const float* __restrict__ P,
                               float* __restrict__ M) {
  int i = blockIdx.x * blockDim.x + threadIdx.x;
  if (i >= NFULL) return;
  M[i] = (S[i] == P[i]) ? 1.f : 0.f;
}

__global__ void suppress_kernel(const float* __restrict__ S, const float* __restrict__ PM,
                                float* __restrict__ SS) {
  int i = blockIdx.x * blockDim.x + threadIdx.x;
  if (i >= NFULL) return;
  SS[i] = (PM[i] > 0.f) ? 0.f : S[i];
}

__global__ void update_mask_kernel(float* __restrict__ M, const float* __restrict__ SS,
                                   const float* __restrict__ PS,
                                   const float* __restrict__ PM) {
  int i = blockIdx.x * blockDim.x + threadIdx.x;
  if (i >= NFULL) return;
  bool m = (M[i] != 0.f);
  bool supp = (PM[i] > 0.f);
  bool nm = (SS[i] == PS[i]);
  M[i] = (m || (nm && !supp)) ? 1.f : 0.f;
}

__global__ void finalize_kernel(const float* __restrict__ S, const float* __restrict__ M,
                                float* __restrict__ out) {
  int i = blockIdx.x * blockDim.x + threadIdx.x;
  if (i >= NFULL) return;
  out[i] = (M[i] != 0.f) ? S[i] : 0.f;
}

static inline int nblk(long long n, int b) { return (int)((n + b - 1) / b); }

extern "C" void kernel_launch(void* const* d_in, const int* in_sizes, int n_in,
                              void* d_out, int out_size, void* d_ws, size_t ws_size,
                              hipStream_t stream) {
  const float* x   = (const float*)d_in[0];
  const float* wPa = (const float*)d_in[1];
  const float* bPa = (const float*)d_in[2];
  const float* wPd = (const float*)d_in[3];
  const float* bPd = (const float*)d_in[4];
  const float* wPg = (const float*)d_in[5];
  const float* bPg = (const float*)d_in[6];

  char* ws = (char*)d_ws;
  const size_t sz_xb  = (size_t)NPIX * 512 * 2;   // 78,643,200
  const size_t sz_h1  = (size_t)NPIX * 256 * 2;   // 39,321,600
  const size_t sz_h2  = (size_t)NPIX * 128 * 2;   // 19,660,800
  const size_t sz_wpa = 9ull * 16 * 16 * 512 * 2; //  2,359,296
  const size_t sz_wpd = 9ull * 8 * 8 * 512 * 2;   //    589,824
  const size_t sz_wpg = 1ull * 4 * 5 * 512 * 2;   //     20,480

  __bf16* xb  = (__bf16*)(ws);
  __bf16* h1  = (__bf16*)(ws + sz_xb);
  __bf16* h2  = (__bf16*)(ws + sz_xb + sz_h1);
  __bf16* wpa = (__bf16*)(ws + sz_xb + sz_h1 + sz_h2);
  __bf16* wpd = (__bf16*)(ws + sz_xb + sz_h1 + sz_h2 + sz_wpa);
  __bf16* wpg = (__bf16*)(ws + sz_xb + sz_h1 + sz_h2 + sz_wpa + sz_wpd);
  float* logits = (float*)(ws + sz_xb + sz_h1 + sz_h2 + sz_wpa + sz_wpd + sz_wpg);

  // NMS planes alias the (dead-by-then) xb/h1 regions: 6 x 19.66MB = 117.96MB
  const size_t sz_pl = (size_t)NFULL * 4;
  float* S  = (float*)(ws + 0 * sz_pl);
  float* T  = (float*)(ws + 1 * sz_pl);
  float* P  = (float*)(ws + 2 * sz_pl);
  float* M  = (float*)(ws + 3 * sz_pl);
  float* SS = (float*)(ws + 4 * sz_pl);
  float* PS = (float*)(ws + 5 * sz_pl);

  // 1) input convert/transpose + weight repack
  {
    long long n = (long long)NPIX * 512;
    nchw_to_nhwc_bf16_kernel<<<nblk(n, 256), 256, 0, stream>>>(x, xb, 512, HC, WC, (int)n);
  }
  pack_weights_kernel<<<nblk(9 * 16 * 16 * 512, 256), 256, 0, stream>>>(
      wPa, wpa, 256, 512, 9, 16, 16, 9 * 16 * 16 * 512);
  pack_weights_kernel<<<nblk(9 * 8 * 8 * 512, 256), 256, 0, stream>>>(
      wPd, wpd, 128, 256, 9, 8, 8, 9 * 8 * 8 * 512);
  pack_weights_kernel<<<nblk(1 * 4 * 5 * 512, 256), 256, 0, stream>>>(
      wPg, wpg, 65, 128, 1, 4, 5, 1 * 4 * 5 * 512);

  // 2) conv stack (WMMA, async double-buffered LDS staging)
  dim3 grid(HC, BATCH);
  conv3x3_wmma<512, 256, 2><<<grid, 256, 0, stream>>>(xb, wpa, bPa, h1);
  conv3x3_wmma<256, 128, 1><<<grid, 256, 0, stream>>>(h1, wpd, bPd, h2);
  conv1x1_wmma<<<grid, 160, 0, stream>>>(h2, wpg, bPg, logits);

  // 3) softmax + pixel unshuffle
  softmax_unshuffle_kernel<<<nblk(NPIX, 256), 256, 0, stream>>>(logits, S);

  // 4) simple_nms (radius 4, two refinement iterations)
  const int nb = nblk(NFULL, 256);
  maxpool_h9_kernel<<<nb, 256, 0, stream>>>(S, T);
  maxpool_v9_kernel<<<nb, 256, 0, stream>>>(T, P);
  eq_mask_kernel<<<nb, 256, 0, stream>>>(S, P, M);
  for (int it = 0; it < 2; ++it) {
    maxpool_h9_kernel<<<nb, 256, 0, stream>>>(M, T);
    maxpool_v9_kernel<<<nb, 256, 0, stream>>>(T, P);      // P = maxpool(mask)
    suppress_kernel<<<nb, 256, 0, stream>>>(S, P, SS);
    maxpool_h9_kernel<<<nb, 256, 0, stream>>>(SS, T);
    maxpool_v9_kernel<<<nb, 256, 0, stream>>>(T, PS);     // PS = maxpool(supp_scores)
    update_mask_kernel<<<nb, 256, 0, stream>>>(M, SS, PS, P);
  }
  finalize_kernel<<<nb, 256, 0, stream>>>(S, M, (float*)d_out);
}
